// MyGATv2_29386166239377
// MI455X (gfx1250) — matrix operand order repaired and checked by reading, not compiled, not run on hardware
//
#include <hip/hip_runtime.h>

typedef __bf16 bf16;
typedef __attribute__((ext_vector_type(8)))  __bf16 v8bf;
typedef __attribute__((ext_vector_type(16))) __bf16 v16bf;
typedef __attribute__((ext_vector_type(8)))  float  v8f;

#define NEG_SLOPE 0.2f

__device__ __forceinline__ bf16 f2bf(float f) {
  unsigned u = __float_as_uint(f);
  unsigned r = (u + 0x7FFFu + ((u >> 16) & 1u)) >> 16;   // round-to-nearest-even
  unsigned short hs = (unsigned short)r;
  bf16 o; __builtin_memcpy(&o, &hs, 2); return o;
}

// order-preserving float<->uint encoding for atomicMax on floats
__device__ __forceinline__ unsigned encf(float f) {
  unsigned u = __float_as_uint(f);
  return (u & 0x80000000u) ? ~u : (u | 0x80000000u);
}
__device__ __forceinline__ float decf(unsigned e) {
  unsigned u = (e & 0x80000000u) ? (e ^ 0x80000000u) : ~e;
  return __uint_as_float(u);
}

__device__ __forceinline__ void edge_sd(const long long* __restrict__ ei, int E, int e,
                                        int& s, int& d) {
  if (e < E) { s = (int)ei[e]; d = (int)ei[(size_t)E + e]; }
  else       { s = e - E; d = s; }           // self loops appended after real edges
}
__device__ __forceinline__ int edge_dst(const long long* __restrict__ ei, int E, int e) {
  return (e < E) ? (int)ei[(size_t)E + e] : (e - E);
}

// gfx1250 async global->LDS copy (ASYNCcnt-tracked), 16B per active lane
__device__ __forceinline__ void async_ld_b128(unsigned lds_off, const void* gptr) {
  asm volatile("global_load_async_to_lds_b128 %0, %1, off"
               :: "v"(lds_off), "v"(gptr) : "memory");
}
__device__ __forceinline__ void wait_async0() {
  asm volatile("s_wait_asynccnt 0x0" ::: "memory");
}

// ---------------- utility kernels ----------------
__global__ void k_f32_to_bf16(const float* __restrict__ in, bf16* __restrict__ out, int n) {
  int i = blockIdx.x * blockDim.x + threadIdx.x;
  if (i < n) out[i] = f2bf(in[i]);
}
__global__ void k_fill_f32(float* __restrict__ p, float v, int n) {
  int i = blockIdx.x * blockDim.x + threadIdx.x;
  if (i < n) p[i] = v;
}
__global__ void k_fill_u32(unsigned* __restrict__ p, unsigned v, int n) {
  int i = blockIdx.x * blockDim.x + threadIdx.x;
  if (i < n) p[i] = v;
}
__global__ void k_bias_act(float* __restrict__ y, const float* __restrict__ b,
                           int HC, int total, int relu) {
  int i = blockIdx.x * blockDim.x + threadIdx.x;
  if (i >= total) return;
  float v = y[i] + b[i % HC];
  if (relu) v = fmaxf(v, 0.f);
  y[i] = v;
}

// ---------------- WMMA GEMM: Y[M,Co] = X[M,K](bf16) @ W[K,Co](bf16), f32 out ----------
// block = 256 threads = 8 waves; block tile 64x64; each wave a 16x32 tile
// (one A fragment feeding two 16x16x32 WMMAs per K-step).
__launch_bounds__(256)
__global__ void k_gemm_bf16(const bf16* __restrict__ X, const bf16* __restrict__ W,
                            float* __restrict__ Y, int M, int K, int Co) {
  __shared__ bf16 As[64][40];     // [row][k], +8 pad (80B row stride, 16B aligned)
  __shared__ bf16 Bs[64][40];     // transposed: [n][k], +8 pad

  const int tid  = threadIdx.x;
  const int lane = tid & 31;
  const int wave = tid >> 5;
  const int rowSub = (wave >> 1) * 16;     // 0,16,32,48
  const int colSub = (wave & 1) * 32;      // 0,32
  const int gRow0 = blockIdx.x * 64;
  const int gCol0 = blockIdx.y * 64;

  const int rA  = tid >> 2;                // 0..63  (A staging row)
  const int k8A = (tid & 3) << 3;          // 0,8,16,24
  const bool aRowOk = (gRow0 + rA) < M;

  // zero-fill LDS rows past M once; async loads never touch them
  if (!aRowOk) {
    bf16 z = f2bf(0.f);
    v8bf vz; for (int j = 0; j < 8; ++j) vz[j] = z;
    *(v8bf*)(&As[rA][k8A]) = vz;
  }

  v8f acc0 = {0.f,0.f,0.f,0.f,0.f,0.f,0.f,0.f};
  v8f acc1 = {0.f,0.f,0.f,0.f,0.f,0.f,0.f,0.f};

  const unsigned aLds = (unsigned)(size_t)(&As[rA][k8A]);  // low 32 bits = LDS offset

  for (int K0 = 0; K0 < K; K0 += 32) {
    // stage A: 64x32 bf16, async global->LDS, one b128 per lane
    if (aRowOk)
      async_ld_b128(aLds, X + (size_t)(gRow0 + rA) * K + K0 + k8A);

    // stage B transposed: Bs[n][k] = W[K0+k][gCol0+n]  (32k x 64n)
    {
      int k  = tid >> 3;            // 0..31
      int n8 = (tid & 7) << 3;      // 0,8,...,56
      v8bf v = *(const v8bf*)(W + (size_t)(K0 + k) * Co + gCol0 + n8);
      for (int j = 0; j < 8; ++j) Bs[n8 + j][k] = v[j];
    }
    wait_async0();
    __syncthreads();

    // A fragment: lanes 0-15 -> K 0..15; lanes 16-31 -> K 16..31
    const int kbA = (lane < 16) ? 0 : 16;
    const bf16* pa = &As[rowSub + (lane & 15)][0];
    v8bf alo = *(const v8bf*)(pa + kbA);
    v8bf ahi = *(const v8bf*)(pa + kbA + 8);
    v16bf afr = __builtin_shufflevector(alo, ahi, 0,1,2,3,4,5,6,7,8,9,10,11,12,13,14,15);

    // B fragments: col = lane&15 (+16); lanes 0-15 -> K 0..15, lanes 16-31 -> K 16..31
    const int kbB = (lane < 16) ? 0 : 16;
    const bf16* pb0 = &Bs[colSub + (lane & 15)][kbB];
    const bf16* pb1 = &Bs[colSub + 16 + (lane & 15)][kbB];
    v8bf b0lo = *(const v8bf*)(pb0);
    v8bf b0hi = *(const v8bf*)(pb0 + 8);
    v16bf bfr0 = __builtin_shufflevector(b0lo, b0hi, 0,1,2,3,4,5,6,7,8,9,10,11,12,13,14,15);
    v8bf b1lo = *(const v8bf*)(pb1);
    v8bf b1hi = *(const v8bf*)(pb1 + 8);
    v16bf bfr1 = __builtin_shufflevector(b1lo, b1hi, 0,1,2,3,4,5,6,7,8,9,10,11,12,13,14,15);

    acc0 = __builtin_amdgcn_wmma_f32_16x16x32_bf16(false, afr, false, bfr0,
                                                   (short)0, acc0, false, false);
    acc1 = __builtin_amdgcn_wmma_f32_16x16x32_bf16(false, afr, false, bfr1,
                                                   (short)0, acc1, false, false);
    __syncthreads();
  }

  // D: VGPR r holds M = r + (lane<16 ? 0 : 8); col = lane&15
  const int mOff = (lane < 16) ? 0 : 8;
  const int col0 = gCol0 + colSub + (lane & 15);
  for (int r = 0; r < 8; ++r) {
    int row = gRow0 + rowSub + mOff + r;
    if (row < M) {
      Y[(size_t)row * Co + col0]      = acc0[r];
      Y[(size_t)row * Co + col0 + 16] = acc1[r];
    }
  }
}

// ---------------- edge pass 1: scores + scatter-max --------------------------
// one wave per edge; lane owns CH contiguous channels (stays inside one head)
template<int CH>
__launch_bounds__(256)
__global__ void k_edge_score_t(const float* __restrict__ xl, const float* __restrict__ xr,
                               const long long* __restrict__ ei, const float* __restrict__ att,
                               float* __restrict__ scores, unsigned* __restrict__ mEnc,
                               int E, int Etot) {
  constexpr int HC = CH * 32;
  constexpr int H  = CH / 2;       // C = 64
  constexpr int G  = 64 / CH;      // lanes per head
  int e = blockIdx.x * 8 + (threadIdx.x >> 5);
  if (e >= Etot) return;
  int lane = threadIdx.x & 31;
  int s, d; edge_sd(ei, E, e, s, d);
  int cb = lane * CH;
  const float* pl = xl + (size_t)s * HC + cb;
  const float* pr = xr + (size_t)d * HC + cb;
  const float* pa = att + cb;
  float part = 0.f;
#pragma unroll
  for (int i = 0; i < CH; ++i) {
    float v = pl[i] + pr[i];
    v = (v > 0.f) ? v : NEG_SLOPE * v;
    part += v * pa[i];
  }
#pragma unroll
  for (int off = G >> 1; off > 0; off >>= 1)
    part += __shfl_xor(part, off, 32);
  if ((lane & (G - 1)) == 0) {
    int h = lane / G;
    scores[(size_t)e * H + h] = part;
    atomicMax(&mEnc[(size_t)d * H + h], encf(part));
  }
}

// ---------------- edge pass 2: exp + scatter-sum denominator -----------------
__global__ void k_edge_exp(const long long* __restrict__ ei,
                           const float* __restrict__ scores, const unsigned* __restrict__ mEnc,
                           float* __restrict__ exbuf, float* __restrict__ den,
                           int E, int Etot, int H) {
  int t = blockIdx.x * blockDim.x + threadIdx.x;
  if (t >= Etot * H) return;
  int e = t / H, h = t - e * H;
  int d = edge_dst(ei, E, e);
  float m  = decf(mEnc[(size_t)d * H + h]);
  float ex = __expf(scores[t] - m);
  exbuf[t] = ex;
  atomicAdd(&den[(size_t)d * H + h], ex);
}

// ---------------- edge pass 3: weighted scatter-add of messages --------------
template<int CH>
__launch_bounds__(256)
__global__ void k_edge_aggr_t(const float* __restrict__ xl, const long long* __restrict__ ei,
                              const float* __restrict__ exbuf, const float* __restrict__ den,
                              float* __restrict__ out, int E, int Etot) {
  constexpr int HC = CH * 32;
  constexpr int H  = CH / 2;
  constexpr int G  = 64 / CH;
  int e = blockIdx.x * 8 + (threadIdx.x >> 5);
  if (e >= Etot) return;
  int lane = threadIdx.x & 31;
  int s, d; edge_sd(ei, E, e, s, d);
  int h  = lane / G;
  float alpha = exbuf[(size_t)e * H + h] / den[(size_t)d * H + h];
  int cb = lane * CH;
  const float* pl = xl + (size_t)s * HC + cb;
  float* po = out + (size_t)d * HC + cb;
  float m[CH];
#pragma unroll
  for (int i = 0; i < CH; ++i) m[i] = alpha * pl[i];   // b128 gather first
#pragma unroll
  for (int i = 0; i < CH; ++i) atomicAdd(&po[i], m[i]);
}

// ---------------- host orchestration -----------------------------------------
static inline size_t alignup256(size_t x) { return (x + 255) & ~(size_t)255; }
static inline int cdiv(int a, int b) { return (a + b - 1) / b; }

extern "C" void kernel_launch(void* const* d_in, const int* in_sizes, int n_in,
                              void* d_out, int out_size, void* d_ws, size_t ws_size,
                              hipStream_t stream) {
  (void)n_in; (void)out_size; (void)ws_size;
  const float*     x  = (const float*)d_in[0];
  const long long* ei = (const long long*)d_in[1];
  const float* Wl[3]  = {(const float*)d_in[2],  (const float*)d_in[6],  (const float*)d_in[10]};
  const float* Wr[3]  = {(const float*)d_in[3],  (const float*)d_in[7],  (const float*)d_in[11]};
  const float* att[3] = {(const float*)d_in[4],  (const float*)d_in[8],  (const float*)d_in[12]};
  const float* bia[3] = {(const float*)d_in[5],  (const float*)d_in[9],  (const float*)d_in[13]};

  const int Nn   = in_sizes[0] / 128;   // 50000
  const int E    = in_sizes[1] / 2;     // 600000
  const int Etot = E + Nn;              // + self loops
  const int Kin[3]   = {128, 256, 256};
  const int HCout[3] = {256, 256, 64};
  const int Hh[3]    = {4, 4, 1};

  // workspace layout
  char* w = (char*)d_ws;
  auto take = [&](size_t bytes) { char* p = w; w += alignup256(bytes); return p; };
  bf16*     hb   = (bf16*)take((size_t)Nn * 256 * sizeof(bf16));   // bf16 layer input
  bf16*     wlb  = (bf16*)take((size_t)256 * 256 * sizeof(bf16));
  bf16*     wrb  = (bf16*)take((size_t)256 * 256 * sizeof(bf16));
  float*    xl   = (float*)take((size_t)Nn * 256 * sizeof(float));
  float*    xr   = (float*)take((size_t)Nn * 256 * sizeof(float));
  float*    accb = (float*)take((size_t)Nn * 256 * sizeof(float)); // h0/h1
  unsigned* mE   = (unsigned*)take((size_t)Nn * 4 * sizeof(unsigned));
  float*    den  = (float*)take((size_t)Nn * 4 * sizeof(float));
  float*    sc   = (float*)take((size_t)Etot * 4 * sizeof(float));
  float*    ex   = (float*)take((size_t)Etot * 4 * sizeof(float));

  const unsigned NEG_INF_ENC = 0x007FFFFFu;   // encf(-inf)

  // layer 0 input -> bf16
  k_f32_to_bf16<<<cdiv(Nn * 128, 256), 256, 0, stream>>>(x, hb, Nn * 128);

  for (int L = 0; L < 3; ++L) {
    const int K = Kin[L], HC = HCout[L], H = Hh[L];
    k_f32_to_bf16<<<cdiv(K * HC, 256), 256, 0, stream>>>(Wl[L], wlb, K * HC);
    k_f32_to_bf16<<<cdiv(K * HC, 256), 256, 0, stream>>>(Wr[L], wrb, K * HC);

    dim3 gg(cdiv(Nn, 64), cdiv(HC, 64));
    k_gemm_bf16<<<gg, 256, 0, stream>>>(hb, wlb, xl, Nn, K, HC);
    k_gemm_bf16<<<gg, 256, 0, stream>>>(hb, wrb, xr, Nn, K, HC);

    float* out = (L == 2) ? (float*)d_out : accb;
    k_fill_u32<<<cdiv(Nn * H, 256), 256, 0, stream>>>(mE, NEG_INF_ENC, Nn * H);
    k_fill_f32<<<cdiv(Nn * H, 256), 256, 0, stream>>>(den, 0.f, Nn * H);
    k_fill_f32<<<cdiv(Nn * HC, 256), 256, 0, stream>>>(out, 0.f, Nn * HC);

    if (H == 4) {
      k_edge_score_t<8><<<cdiv(Etot, 8), 256, 0, stream>>>(xl, xr, ei, att[L], sc, mE, E, Etot);
    } else {
      k_edge_score_t<2><<<cdiv(Etot, 8), 256, 0, stream>>>(xl, xr, ei, att[L], sc, mE, E, Etot);
    }
    k_edge_exp<<<cdiv(Etot * H, 256), 256, 0, stream>>>(ei, sc, mE, ex, den, E, Etot, H);
    if (H == 4) {
      k_edge_aggr_t<8><<<cdiv(Etot, 8), 256, 0, stream>>>(xl, ei, ex, den, out, E, Etot);
    } else {
      k_edge_aggr_t<2><<<cdiv(Etot, 8), 256, 0, stream>>>(xl, ei, ex, den, out, E, Etot);
    }

    k_bias_act<<<cdiv(Nn * HC, 256), 256, 0, stream>>>(out, bia[L], HC, Nn * HC, (L < 2) ? 1 : 0);
    if (L < 2)
      k_f32_to_bf16<<<cdiv(Nn * HC, 256), 256, 0, stream>>>(accb, hb, Nn * HC);
  }
}
